// VRNN_14937896255664
// MI455X (gfx1250) — matrix operand order repaired and checked by reading, not compiled, not run on hardware
//
#include <hip/hip_runtime.h>
#include <hip/hip_bf16.h>
#include <math.h>

// ---------------- dims ----------------
#define T_STEPS 200
#define BATCH   128
#define XDIM    784
#define HDIM    512
#define ZDIM    128

#define EPSV 1.1920929e-07f

// ---------------- WMMA types ----------------
typedef __bf16 bf16;
typedef bf16  v16bf __attribute__((ext_vector_type(16)));
typedef bf16  v8bf  __attribute__((ext_vector_type(8)));
typedef float v8f   __attribute__((ext_vector_type(8)));

__device__ __forceinline__ v8f vzero() {
  v8f z;
#pragma unroll
  for (int i = 0; i < 8; ++i) z[i] = 0.f;
  return z;
}

__device__ __forceinline__ v8f wmma_bf16(v8f acc, v16bf a, v16bf b) {
  return __builtin_amdgcn_wmma_f32_16x16x32_bf16(false, a, false, b, (short)0, acc, false, false);
}

// A fragment (16x32 bf16) from LDS, row-major act[16][pitch].
// lane<16 -> K {kb+0..7, kb+16..23}; lane>=16 -> K {kb+8..15, kb+24..31}; m = lane&15.
__device__ __forceinline__ v16bf load_a_frag(const bf16* A, int pitch, int kbase, int lane) {
  const int m   = lane & 15;
  const int off = m * pitch + kbase + ((lane >> 4) << 3);
  v8bf lo = *reinterpret_cast<const v8bf*>(A + off);
  v8bf hi = *reinterpret_cast<const v8bf*>(A + off + 16);
  v16bf r;
#pragma unroll
  for (int j = 0; j < 8; ++j) { r[j] = lo[j]; r[j + 8] = hi[j]; }
  return r;
}

// B fragment pre-swizzled: frag = 32 lanes x 16 bf16 contiguous (1KB).
__device__ __forceinline__ v16bf load_b_frag(const bf16* Wf, int lane) {
  return *reinterpret_cast<const v16bf*>(Wf + lane * 16);
}

// ---- 1-tile GEMM, even/odd pair rotation (no register copies) ----
__device__ __forceinline__ v8f gemm_seg(v8f acc, const bf16* A, int pitch,
                                        const bf16* Wf, int nk, int lane) {
  int k = 0;
  if (nk & 1) {
    v16bf a = load_a_frag(A, pitch, 0, lane);
    acc = wmma_bf16(acc, a, load_b_frag(Wf, lane));
    k = 1;
  }
  if (k >= nk) return acc;
  v16bf a0  = load_a_frag(A, pitch, k * 32, lane);
  v16bf a1  = load_a_frag(A, pitch, (k + 1) * 32, lane);
  v16bf b0  = load_b_frag(Wf + (size_t)k * 512, lane);
  v16bf b1  = load_b_frag(Wf + (size_t)(k + 1) * 512, lane);
#pragma unroll 1
  for (; k + 2 < nk; k += 2) {
    __builtin_prefetch((const void*)(Wf + (size_t)(k + 4) * 512 + (size_t)lane * 16), 0, 1);
    acc = wmma_bf16(acc, a0, b0);
    a0 = load_a_frag(A, pitch, (k + 2) * 32, lane);
    b0 = load_b_frag(Wf + (size_t)(k + 2) * 512, lane);
    acc = wmma_bf16(acc, a1, b1);
    a1 = load_a_frag(A, pitch, (k + 3) * 32, lane);
    b1 = load_b_frag(Wf + (size_t)(k + 3) * 512, lane);
  }
  acc = wmma_bf16(acc, a0, b0);
  acc = wmma_bf16(acc, a1, b1);
  return acc;
}

// ---- 2-tile GEMM sharing the A stream (2 WMMA per A fragment) ----
__device__ __forceinline__ void gemm2_seg(v8f& c0, v8f& c1,
                                          const bf16* A, int pitch,
                                          const bf16* W0, const bf16* W1,
                                          int nk, int lane) {
  int k = 0;
  if (nk & 1) {
    v16bf a = load_a_frag(A, pitch, 0, lane);
    c0 = wmma_bf16(c0, a, load_b_frag(W0, lane));
    c1 = wmma_bf16(c1, a, load_b_frag(W1, lane));
    k = 1;
  }
  if (k >= nk) return;
  v16bf a0  = load_a_frag(A, pitch, k * 32, lane);
  v16bf a1  = load_a_frag(A, pitch, (k + 1) * 32, lane);
  v16bf b00 = load_b_frag(W0 + (size_t)k * 512, lane);
  v16bf b10 = load_b_frag(W1 + (size_t)k * 512, lane);
  v16bf b01 = load_b_frag(W0 + (size_t)(k + 1) * 512, lane);
  v16bf b11 = load_b_frag(W1 + (size_t)(k + 1) * 512, lane);
#pragma unroll 1
  for (; k + 2 < nk; k += 2) {
    __builtin_prefetch((const void*)(W0 + (size_t)(k + 4) * 512 + (size_t)lane * 16), 0, 1);
    __builtin_prefetch((const void*)(W1 + (size_t)(k + 4) * 512 + (size_t)lane * 16), 0, 1);
    c0 = wmma_bf16(c0, a0, b00);
    c1 = wmma_bf16(c1, a0, b10);
    a0  = load_a_frag(A, pitch, (k + 2) * 32, lane);
    b00 = load_b_frag(W0 + (size_t)(k + 2) * 512, lane);
    b10 = load_b_frag(W1 + (size_t)(k + 2) * 512, lane);
    c0 = wmma_bf16(c0, a1, b01);
    c1 = wmma_bf16(c1, a1, b11);
    a1  = load_a_frag(A, pitch, (k + 3) * 32, lane);
    b01 = load_b_frag(W0 + (size_t)(k + 3) * 512, lane);
    b11 = load_b_frag(W1 + (size_t)(k + 3) * 512, lane);
  }
  c0 = wmma_bf16(c0, a0, b00);
  c1 = wmma_bf16(c1, a0, b10);
  c0 = wmma_bf16(c0, a1, b01);
  c1 = wmma_bf16(c1, a1, b11);
}

// ---- 3-tile GEMM sharing the A stream (GRU gates r,u,n) ----
__device__ __forceinline__ void gemm3_seg(v8f& c0, v8f& c1, v8f& c2,
                                          const bf16* A, int pitch,
                                          const bf16* W0, const bf16* W1, const bf16* W2,
                                          int nk, int lane) {
  int k = 0;
  if (nk & 1) {
    v16bf a = load_a_frag(A, pitch, 0, lane);
    c0 = wmma_bf16(c0, a, load_b_frag(W0, lane));
    c1 = wmma_bf16(c1, a, load_b_frag(W1, lane));
    c2 = wmma_bf16(c2, a, load_b_frag(W2, lane));
    k = 1;
  }
  if (k >= nk) return;
  v16bf a0  = load_a_frag(A, pitch, k * 32, lane);
  v16bf a1  = load_a_frag(A, pitch, (k + 1) * 32, lane);
  v16bf b00 = load_b_frag(W0 + (size_t)k * 512, lane);
  v16bf b10 = load_b_frag(W1 + (size_t)k * 512, lane);
  v16bf b20 = load_b_frag(W2 + (size_t)k * 512, lane);
  v16bf b01 = load_b_frag(W0 + (size_t)(k + 1) * 512, lane);
  v16bf b11 = load_b_frag(W1 + (size_t)(k + 1) * 512, lane);
  v16bf b21 = load_b_frag(W2 + (size_t)(k + 1) * 512, lane);
#pragma unroll 1
  for (; k + 2 < nk; k += 2) {
    __builtin_prefetch((const void*)(W0 + (size_t)(k + 4) * 512 + (size_t)lane * 16), 0, 1);
    __builtin_prefetch((const void*)(W1 + (size_t)(k + 4) * 512 + (size_t)lane * 16), 0, 1);
    __builtin_prefetch((const void*)(W2 + (size_t)(k + 4) * 512 + (size_t)lane * 16), 0, 1);
    c0 = wmma_bf16(c0, a0, b00);
    c1 = wmma_bf16(c1, a0, b10);
    c2 = wmma_bf16(c2, a0, b20);
    a0  = load_a_frag(A, pitch, (k + 2) * 32, lane);
    b00 = load_b_frag(W0 + (size_t)(k + 2) * 512, lane);
    b10 = load_b_frag(W1 + (size_t)(k + 2) * 512, lane);
    b20 = load_b_frag(W2 + (size_t)(k + 2) * 512, lane);
    c0 = wmma_bf16(c0, a1, b01);
    c1 = wmma_bf16(c1, a1, b11);
    c2 = wmma_bf16(c2, a1, b21);
    a1  = load_a_frag(A, pitch, (k + 3) * 32, lane);
    b01 = load_b_frag(W0 + (size_t)(k + 3) * 512, lane);
    b11 = load_b_frag(W1 + (size_t)(k + 3) * 512, lane);
    b21 = load_b_frag(W2 + (size_t)(k + 3) * 512, lane);
  }
  c0 = wmma_bf16(c0, a0, b00);
  c1 = wmma_bf16(c1, a0, b10);
  c2 = wmma_bf16(c2, a0, b20);
  c0 = wmma_bf16(c0, a1, b01);
  c1 = wmma_bf16(c1, a1, b11);
  c2 = wmma_bf16(c2, a1, b21);
}

// ---------------- pre-swizzled weight arena (bf16 elements in d_ws) ----------------
#define WOFF_PHIX1 0u        // 512x800  KF=25
#define WOFF_PHIX2 409600u   // 512x512  KF=16
#define WOFF_PHIZ  671744u   // 512x128  KF=4
#define WOFF_ENC1  737280u   // 512x1024 KF=32
#define WOFF_ENC2  1261568u  // 512x512
#define WOFF_ENCM  1523712u  // 128x512
#define WOFF_ENCS  1589248u  // 128x512
#define WOFF_PRIOR 1654784u  // 512x512
#define WOFF_PRIM  1916928u  // 128x512
#define WOFF_PRIS  1982464u  // 128x512
#define WOFF_DEC1  2048000u  // 512x1024
#define WOFF_DEC2  2572288u  // 512x512
#define WOFF_DECM  2834432u  // 784x512
#define WOFF_DECS  3235840u  // 784x512
#define WOFF_WIH   3637248u  // 1536x1024
#define WOFF_WHH   5210112u  // 1536x512
#define WTOTAL     5996544u
#define PARTIALS_BYTE_OFF (12u << 20)

struct WSrcs  { const float* p[16]; };
struct Biases {
  const float *phi_x_b1, *phi_x_b2, *phi_z_b, *enc_b1, *enc_b2,
              *enc_mean_b, *enc_std_b, *prior_b, *prior_mean_b, *prior_std_b,
              *dec_b1, *dec_b2, *dec_mean_b, *dec_std_b;
};

// fp32 [O][I] -> bf16 B-fragments: frag(nt,kt)[lane][j] = W[nt*16+(lane&15)][kt*32+((lane>=16)?8:0)+(j&7)+((j>>3)<<4)]
__global__ void prep_weights(WSrcs src, bf16* wf) {
  const size_t OFFS[17] = {0u, WOFF_PHIX2, WOFF_PHIZ, WOFF_ENC1, WOFF_ENC2, WOFF_ENCM,
                           WOFF_ENCS, WOFF_PRIOR, WOFF_PRIM, WOFF_PRIS, WOFF_DEC1,
                           WOFF_DEC2, WOFF_DECM, WOFF_DECS, WOFF_WIH, WOFF_WHH, WTOTAL};
  const int Iarr[16]  = {784, 512, 128, 1024, 512, 512, 512, 512, 512, 512, 1024, 512, 512, 512, 1024, 512};
  const int IParr[16] = {800, 512, 128, 1024, 512, 512, 512, 512, 512, 512, 1024, 512, 512, 512, 1024, 512};

  const size_t ngroups = WTOTAL / 16u;
  for (size_t g = (size_t)blockIdx.x * blockDim.x + threadIdx.x; g < ngroups;
       g += (size_t)gridDim.x * blockDim.x) {
    size_t e0 = g * 16u;
    int l = 0;
    while (e0 >= OFFS[l + 1]) ++l;
    size_t local = e0 - OFFS[l];
    const int KF = IParr[l] >> 5;
    const int f  = (int)(local >> 9);
    const int L  = (int)((local >> 4) & 31u);
    const int nt = f / KF, kt = f - nt * KF;
    const int n  = nt * 16 + (L & 15);
    const int kb = kt * 32 + ((L >> 4) << 3);
    const float* W = src.p[l];
    const int I = Iarr[l];
#pragma unroll
    for (int j = 0; j < 16; ++j) {
      int k = kb + (j & 7) + ((j >> 3) << 4);
      float v = (k < I) ? W[(size_t)n * I + k] : 0.f;
      wf[e0 + j] = (bf16)v;
    }
  }
}

// ---------------- LDS layout (dynamic, ~165KB -> one WG per WGP) ----------------
#define XP 808
#define HP 520
#define ZP 136
#define SM_XB   0u
#define SM_HB0  25856u
#define SM_HB1  42496u
#define SM_A1   59136u
#define SM_A2   75776u   // phi_x (persists)
#define SM_A3   92416u   // enc_h / dec_h
#define SM_A4   109056u  // phi_z (persists)
#define SM_ZB   125696u
#define SM_EMF  130048u
#define SM_ESF  138240u
#define SM_PMF  146432u
#define SM_PSF  154624u
#define SM_RED  162816u
#define SMEM_BYTES 164864u

__device__ __forceinline__ float sigmoidf_(float x) { return 1.f / (1.f + expf(-x)); }
__device__ __forceinline__ float softplusf_(float x) { return x > 20.f ? x : log1pf(expf(x)); }

// relu( A1@W[:, :32*nk1].T (+ A2@W[:, 32*nk1:].T) + b ) into dst[16][HP].
// 2 tiles/wave (t = wave, wave+16) computed together sharing the A stream.
__device__ __forceinline__ void dense512(const bf16* A1, int p1, int nk1,
                                         const bf16* A2, int nk2,
                                         const bf16* Wbase, int KF, const float* bias,
                                         bf16* dst, int wave, int lane) {
  const bf16* W0 = Wbase + (size_t)wave * KF * 512;
  const bf16* W1 = Wbase + (size_t)(wave + 16) * KF * 512;
  v8f c0 = vzero(), c1 = vzero();
  gemm2_seg(c0, c1, A1, p1, W0, W1, nk1, lane);
  if (A2) gemm2_seg(c0, c1, A2, HP, W0 + (size_t)nk1 * 512, W1 + (size_t)nk1 * 512, nk2, lane);
  const int ln = lane & 15;
  const int mb = (lane >> 4) << 3;
  const int n0 = wave * 16 + ln;
  const int n1 = (wave + 16) * 16 + ln;
  const float bv0 = bias[n0], bv1 = bias[n1];
#pragma unroll
  for (int r = 0; r < 8; ++r) {
    dst[(mb + r) * HP + n0] = (bf16)fmaxf(c0[r] + bv0, 0.f);
    dst[(mb + r) * HP + n1] = (bf16)fmaxf(c1[r] + bv1, 0.f);
  }
}

// mean/std heads (N=128 each): waves 0-7 -> mean, 8-15 -> softplus(std)
__device__ __forceinline__ void heads128(const bf16* A, const bf16* Wm, const bf16* Ws,
                                         const float* bm, const float* bs,
                                         float* mf, float* sf, float* gm, float* gs,
                                         int wave, int lane) {
  const bool isStd = wave >= 8;
  const int t = wave & 7;
  const bf16* W = (isStd ? Ws : Wm) + (size_t)t * 16 * 512;
  const float* bias = isStd ? bs : bm;
  v8f acc = gemm_seg(vzero(), A, HP, W, 16, lane);
  const int n = t * 16 + (lane & 15);
  const int mb = (lane >> 4) << 3;
  const float bv = bias[n];
  float* f = isStd ? sf : mf;
  float* g = isStd ? gs : gm;
#pragma unroll
  for (int r = 0; r < 8; ++r) {
    const int m = mb + r;
    float v = acc[r] + bv;
    if (isStd) v = softplusf_(v);
    f[m * ZDIM + n] = v;
    if (g) g[(size_t)m * ZDIM + n] = v;
  }
}

__global__ void __launch_bounds__(512)
vrnn_kernel(const float* __restrict__ xg, const float* __restrict__ epsg,
            Biases B, const bf16* __restrict__ wf, float* __restrict__ out,
            float* __restrict__ partials) {
  extern __shared__ char smem[];
  bf16*  xb  = (bf16*)(smem + SM_XB);
  bf16*  hb0 = (bf16*)(smem + SM_HB0);
  bf16*  hb1 = (bf16*)(smem + SM_HB1);
  bf16*  a1  = (bf16*)(smem + SM_A1);
  bf16*  a2  = (bf16*)(smem + SM_A2);
  bf16*  a3  = (bf16*)(smem + SM_A3);
  bf16*  a4  = (bf16*)(smem + SM_A4);
  bf16*  zb  = (bf16*)(smem + SM_ZB);
  float* emf = (float*)(smem + SM_EMF);
  float* esf = (float*)(smem + SM_ESF);
  float* pmf = (float*)(smem + SM_PMF);
  float* psf = (float*)(smem + SM_PSF);
  float* red = (float*)(smem + SM_RED);

  const int tid  = threadIdx.x;
  const int lane = tid & 31;
  const int wave = tid >> 5;
  const int wg   = blockIdx.x;   // 8 blocks, 16 batch rows each (independent h)
  const int row0 = wg * 16;

  float* out_ems = out + 2;
  float* out_ess = out + 2 + 3276800;
  float* out_dms = out + 2 + 6553600;
  float* out_dss = out + 2 + 26624000;

  for (int i = tid; i < 16 * HP; i += 512) hb0[i] = (bf16)0.f;
  for (int i = tid; i < 16 * 24; i += 512) {
    int m = i / 24, c = 784 + (i - m * 24);
    xb[m * XP + c] = (bf16)0.f;
  }
  float kacc = 0.f, nacc = 0.f;
  bf16* hcur = hb0;
  bf16* hnxt = hb1;
  __syncthreads();

#pragma unroll 1
  for (int t = 0; t < T_STEPS; ++t) {
    // x_t -> LDS (bf16)
    for (int i = tid; i < 16 * XDIM; i += 512) {
      int m = i / XDIM, c = i - m * XDIM;
      xb[m * XP + c] = (bf16)xg[((size_t)t * BATCH + row0 + m) * XDIM + c];
    }
    __syncthreads();
    // phi_x MLP
    dense512(xb, XP, 25, nullptr, 0, wf + WOFF_PHIX1, 25, B.phi_x_b1, a1, wave, lane);
    __syncthreads();
    dense512(a1, HP, 16, nullptr, 0, wf + WOFF_PHIX2, 16, B.phi_x_b2, a2, wave, lane);
    __syncthreads();
    // encoder: [phi_x | h]
    dense512(a2, HP, 16, hcur, 16, wf + WOFF_ENC1, 32, B.enc_b1, a1, wave, lane);
    __syncthreads();
    dense512(a1, HP, 16, nullptr, 0, wf + WOFF_ENC2, 16, B.enc_b2, a3, wave, lane);
    __syncthreads();
    // enc mean/std heads + prior (disjoint src/dst -> single barrier phase)
    heads128(a3, wf + WOFF_ENCM, wf + WOFF_ENCS, B.enc_mean_b, B.enc_std_b,
             emf, esf,
             out_ems + ((size_t)t * BATCH + row0) * ZDIM,
             out_ess + ((size_t)t * BATCH + row0) * ZDIM, wave, lane);
    dense512(hcur, HP, 16, nullptr, 0, wf + WOFF_PRIOR, 16, B.prior_b, a1, wave, lane);
    __syncthreads();
    heads128(a1, wf + WOFF_PRIM, wf + WOFF_PRIS, B.prior_mean_b, B.prior_std_b,
             pmf, psf, nullptr, nullptr, wave, lane);
    __syncthreads();
    // z = eps*es + em ; KLD accum
    for (int i = tid; i < 16 * ZDIM; i += 512) {
      int m = i >> 7, n = i & 127;
      float em = emf[i], es = esf[i], pm = pmf[i], ps = psf[i];
      float e = epsg[((size_t)t * BATCH + row0 + m) * ZDIM + n];
      zb[m * ZP + n] = (bf16)(e * es + em);
      float d = em - pm;
      kacc += 2.f * logf(ps + EPSV) - 2.f * logf(es + EPSV)
              + (es * es + d * d) / (ps * ps) - 1.f;
    }
    __syncthreads();
    // phi_z
    dense512(zb, ZP, 4, nullptr, 0, wf + WOFF_PHIZ, 4, B.phi_z_b, a4, wave, lane);
    __syncthreads();
    // decoder: [phi_z | h]
    dense512(a4, HP, 16, hcur, 16, wf + WOFF_DEC1, 32, B.dec_b1, a1, wave, lane);
    __syncthreads();
    dense512(a1, HP, 16, nullptr, 0, wf + WOFF_DEC2, 16, B.dec_b2, a3, wave, lane);
    __syncthreads();
    // dec heads N=784 (49 tiles): mean+std tiles share the A stream
#pragma unroll 1
    for (int tt = 0; tt < 4; ++tt) {
      const int tl = wave + tt * 16;
      if (tl < 49) {
        const int n = tl * 16 + (lane & 15);
        const int mb = (lane >> 4) << 3;
        v8f am = vzero(), as_ = vzero();
        gemm2_seg(am, as_, a3, HP,
                  wf + WOFF_DECM + (size_t)tl * 16 * 512,
                  wf + WOFF_DECS + (size_t)tl * 16 * 512, 16, lane);
        const float bm = B.dec_mean_b[n];
        const float bs = B.dec_std_b[n];
#pragma unroll
        for (int r = 0; r < 8; ++r) {
          const int m = mb + r;
          const size_t gi = ((size_t)t * BATCH + row0 + m) * XDIM + n;
          float dm = sigmoidf_(am[r] + bm);
          out_dms[gi] = dm;
          float xv = xg[gi];
          nacc -= xv * logf(dm + EPSV) + (1.f - xv) * logf(1.f - dm - EPSV);
          out_dss[gi] = softplusf_(as_[r] + bs);
        }
      }
    }
    // GRU: 3 gate columns share each A stream (phi_x, phi_z, h). Reads hcur, writes hnxt.
#pragma unroll 1
    for (int tt = 0; tt < 2; ++tt) {
      const int tl = wave + tt * 16;
      const bf16* Wih = wf + WOFF_WIH;
      const bf16* Whh = wf + WOFF_WHH;
      v8f rs = vzero(), us = vzero(), gn = vzero();
      gemm3_seg(rs, us, gn, a2, HP,
                Wih + ((size_t)tl * 32) * 512,
                Wih + ((size_t)(32 + tl) * 32) * 512,
                Wih + ((size_t)(64 + tl) * 32) * 512, 16, lane);
      gemm3_seg(rs, us, gn, a4, HP,
                Wih + ((size_t)tl * 32 + 16) * 512,
                Wih + ((size_t)(32 + tl) * 32 + 16) * 512,
                Wih + ((size_t)(64 + tl) * 32 + 16) * 512, 16, lane);
      v8f hr = vzero(), hu = vzero(), hn = vzero();
      gemm3_seg(hr, hu, hn, hcur, HP,
                Whh + ((size_t)tl * 16) * 512,
                Whh + ((size_t)(32 + tl) * 16) * 512,
                Whh + ((size_t)(64 + tl) * 16) * 512, 16, lane);
      rs = rs + hr;
      us = us + hu;
      const int n = tl * 16 + (lane & 15);
      const int mb = (lane >> 4) << 3;
#pragma unroll
      for (int r = 0; r < 8; ++r) {
        const int m = mb + r;
        float rr = sigmoidf_(rs[r]);
        float uu = sigmoidf_(us[r]);
        float nn = tanhf(gn[r] + rr * hn[r]);
        float hold = (float)hcur[m * HP + n];
        hnxt[m * HP + n] = (bf16)((1.f - uu) * nn + uu * hold);
      }
    }
    __syncthreads();
    bf16* tmp = hcur; hcur = hnxt; hnxt = tmp;  // uniform swap
  }

  // deterministic in-block reduction, fixed order
  red[tid] = 0.5f * kacc;
  __syncthreads();
  if (tid == 0) { float s = 0.f; for (int i = 0; i < 512; ++i) s += red[i]; partials[wg * 2] = s; }
  __syncthreads();
  red[tid] = nacc;
  __syncthreads();
  if (tid == 0) { float s = 0.f; for (int i = 0; i < 512; ++i) s += red[i]; partials[wg * 2 + 1] = s; }
}

__global__ void finalize_kernel(float* out, const float* partials) {
  float k = 0.f, n = 0.f;
  for (int i = 0; i < 8; ++i) { k += partials[i * 2]; n += partials[i * 2 + 1]; }
  out[0] = k;
  out[1] = n;
}

extern "C" void kernel_launch(void* const* d_in, const int* in_sizes, int n_in,
                              void* d_out, int out_size, void* d_ws, size_t ws_size,
                              hipStream_t stream) {
  (void)in_sizes; (void)n_in; (void)out_size; (void)ws_size;
  const float* xg   = (const float*)d_in[0];
  const float* epsg = (const float*)d_in[1];

  WSrcs ws;
  const int widx[16] = {2, 4, 6, 8, 10, 12, 14, 16, 18, 20, 22, 24, 26, 28, 30, 31};
  for (int i = 0; i < 16; ++i) ws.p[i] = (const float*)d_in[widx[i]];

  Biases B;
  B.phi_x_b1 = (const float*)d_in[3];   B.phi_x_b2 = (const float*)d_in[5];
  B.phi_z_b = (const float*)d_in[7];
  B.enc_b1 = (const float*)d_in[9];     B.enc_b2 = (const float*)d_in[11];
  B.enc_mean_b = (const float*)d_in[13]; B.enc_std_b = (const float*)d_in[15];
  B.prior_b = (const float*)d_in[17];
  B.prior_mean_b = (const float*)d_in[19]; B.prior_std_b = (const float*)d_in[21];
  B.dec_b1 = (const float*)d_in[23];    B.dec_b2 = (const float*)d_in[25];
  B.dec_mean_b = (const float*)d_in[27]; B.dec_std_b = (const float*)d_in[29];

  bf16* wf = (bf16*)d_ws;
  float* partials = (float*)((char*)d_ws + PARTIALS_BYTE_OFF);

  prep_weights<<<512, 256, 0, stream>>>(ws, wf);
  vrnn_kernel<<<8, 512, SMEM_BYTES, stream>>>(xg, epsg, B, wf, (float*)d_out, partials);
  finalize_kernel<<<1, 1, 0, stream>>>((float*)d_out, partials);
}
